// MambaNLI_58669253263588
// MI455X (gfx1250) — matrix operand order repaired and checked by reading, not compiled, not run on hardware
//
#include <hip/hip_runtime.h>

// ---------------- problem constants ----------------
#define BSZ    8
#define SEQL   2048
#define DMODEL 256
#define NCAT   3
#define DHID   768            // D
#define EDIM   1536           // ED
#define NSTATE 16             // DS
#define DRANK  48             // DR
#define DRPAD  64             // DR padded to mult of 32 for WMMA K
#define DCONV  4
#define NLAYER 2
#define NLAB   5
#define MROWS  (BSZ*SEQL)     // 16384
#define DBCW   (DRANK + 2*NSTATE)   // 80

typedef __attribute__((ext_vector_type(16))) __bf16 bf16x16;
typedef __attribute__((ext_vector_type(8)))  __bf16 bf16x8;
typedef __attribute__((ext_vector_type(8)))  float  f32x8;

// ---- WMMA GEMM: C[M,N] = A[M,K] * W[N,K]^T, NT 16-wide N-tiles per wave ----
// A-fragment loaded once per K-step and reused for NT WMMAs (register tiling).
template<int NT>
__global__ __launch_bounds__(256)
void k_gemm(const __bf16* __restrict__ A, const __bf16* __restrict__ W,
            float* __restrict__ C, int M, int N, int K) {
  const int lane  = threadIdx.x & 31;
  const int wave  = threadIdx.x >> 5;
  const int nBase = blockIdx.x * (16 * NT);
  const int mTile = ((blockIdx.y << 3) + wave) << 4;
  if (mTile >= M) return;                 // wave-uniform: EXEC stays all-ones
  const int half = lane >> 4;
  const int lr   = lane & 15;

  const __bf16* Ap = A + (size_t)(mTile + lr) * K + (half << 3);
  const __bf16* Wp[NT];
#pragma unroll
  for (int j = 0; j < NT; ++j)
    Wp[j] = W + (size_t)(nBase + j * 16 + lr) * K + (half << 3);

  f32x8 acc[NT];
#pragma unroll
  for (int j = 0; j < NT; ++j) acc[j] = (f32x8){0.f,0.f,0.f,0.f,0.f,0.f,0.f,0.f};

  for (int k = 0; k < K; k += 32) {
    bf16x8 a0 = *(const bf16x8*)(Ap + k);
    bf16x8 a1 = *(const bf16x8*)(Ap + k + 16);
    __builtin_prefetch(Ap + k + 32, 0, 0);       // global_prefetch_b8
    bf16x16 am;
#pragma unroll
    for (int i = 0; i < 8; ++i) { am[i] = a0[i]; am[i + 8] = a1[i]; }
#pragma unroll
    for (int j = 0; j < NT; ++j) {
      bf16x8 b0 = *(const bf16x8*)(Wp[j] + k);
      bf16x8 b1 = *(const bf16x8*)(Wp[j] + k + 16);
      bf16x16 bm;
#pragma unroll
      for (int i = 0; i < 8; ++i) { bm[i] = b0[i]; bm[i + 8] = b1[i]; }
      acc[j] = __builtin_amdgcn_wmma_f32_16x16x32_bf16(
          false, am, false, bm, (short)0, acc[j], false, false);
    }
  }
#pragma unroll
  for (int j = 0; j < NT; ++j) {
    float* Cp = C + (size_t)(mTile + (half << 3)) * N + nBase + j * 16 + lr;
#pragma unroll
    for (int v = 0; v < 8; ++v) Cp[(size_t)v * N] = acc[j][v];
  }
}

// ---------------- elementwise / staging kernels ----------------
__global__ void k_embed(const int* __restrict__ tok, const float* __restrict__ emb,
                        float* __restrict__ x, long n) {
  long i = (long)blockIdx.x * blockDim.x + threadIdx.x;
  if (i >= n) return;
  int  d  = (int)(i % DMODEL);
  long rc = i / DMODEL;
  int  id = tok[rc];
  x[i] = emb[(long)id * DMODEL + d];
}

__global__ void k_f32_to_bf16(const float* __restrict__ in, __bf16* __restrict__ out, long n) {
  long i = (long)blockIdx.x * blockDim.x + threadIdx.x;
  if (i < n) out[i] = (__bf16)in[i];
}

__global__ void k_cvt_pad(const float* __restrict__ in, __bf16* __restrict__ out,
                          long rows, int in_stride, int kin, int kout) {
  long i = (long)blockIdx.x * blockDim.x + threadIdx.x;
  long n = rows * (long)kout;
  if (i >= n) return;
  int  k = (int)(i % kout);
  long r = i / kout;
  out[i] = (k < kin) ? (__bf16)in[r * in_stride + k] : (__bf16)0.0f;
}

__global__ __launch_bounds__(256)
void k_rmsnorm(const float* __restrict__ x, const float* __restrict__ w,
               __bf16* __restrict__ obf, float* __restrict__ of) {
  __shared__ float red[256];
  long row = blockIdx.x;
  const float* xr = x + row * DHID;
  float s = 0.f;
  for (int i = threadIdx.x; i < DHID; i += 256) { float v = xr[i]; s += v * v; }
  red[threadIdx.x] = s; __syncthreads();
  for (int st = 128; st; st >>= 1) {
    if (threadIdx.x < st) red[threadIdx.x] += red[threadIdx.x + st];
    __syncthreads();
  }
  float inv = rsqrtf(red[0] / (float)DHID + 1e-5f);
  for (int i = threadIdx.x; i < DHID; i += 256) {
    float v = xr[i] * inv * w[i];
    if (obf) obf[row * DHID + i] = (__bf16)v;
    if (of)  of[row * DHID + i]  = v;
  }
}

__global__ void k_conv_silu(const float* __restrict__ xz, const float* __restrict__ cw,
                            const float* __restrict__ cb, float* __restrict__ xxs,
                            __bf16* __restrict__ xxsb, long n) {
  long i = (long)blockIdx.x * blockDim.x + threadIdx.x;
  if (i >= n) return;
  int  e   = (int)(i % EDIM);
  long row = i / EDIM;
  int  t   = (int)(row % SEQL);
  float acc = cb[e];
#pragma unroll
  for (int j = 0; j < DCONV; ++j) {
    int ts = t - (DCONV - 1) + j;
    if (ts >= 0) acc += cw[e * DCONV + j] * xz[(row - (DCONV - 1) + j) * (2 * EDIM) + e];
  }
  float s = acc / (1.f + __expf(-acc));
  xxs[i]  = s;
  xxsb[i] = (__bf16)s;
}

__global__ void k_softplus_bias(float* __restrict__ dta, const float* __restrict__ dtb, long n) {
  long i = (long)blockIdx.x * blockDim.x + threadIdx.x;
  if (i >= n) return;
  int e = (int)(i % EDIM);
  float v = dta[i] + dtb[e];
  dta[i] = (v > 20.f) ? v : log1pf(__expf(v));
}

// ---- selective scan: state in registers, B/C double-buffered in LDS via
// ---- global_load_async_to_lds_b128 (ASYNCcnt) prefetching timestep t+1 ----
__global__ __launch_bounds__(256)
void k_scan(const float* __restrict__ u, const float* __restrict__ dta,
            const float* __restrict__ dbc, const float* __restrict__ A_log,
            const float* __restrict__ Dp, float* __restrict__ y) {
  int e = blockIdx.x * 256 + threadIdx.x;   // channel in [0,ED)
  int b = blockIdx.y;
  int tid = threadIdx.x;
  float A[NSTATE], h[NSTATE];
#pragma unroll
  for (int n = 0; n < NSTATE; ++n) { A[n] = -__expf(A_log[e * NSTATE + n]); h[n] = 0.f; }
  float dpar = Dp[e];

  __shared__ float bc[2][2 * NSTATE];       // [buf][B(16) | C(16)]

  // prologue: async-stage B/C of t=0 into buffer 0 (lanes 0-7, 16B each)
  if (tid < 8) {
    const float* g = dbc + ((size_t)b * SEQL + 0) * DBCW + DRANK + tid * 4;
    unsigned lo = ((unsigned)(uintptr_t)&bc[0][0]) + tid * 16;
    asm volatile("global_load_async_to_lds_b128 %0, %1, off" :: "v"(lo), "v"(g) : "memory");
  }

  for (int t = 0; t < SEQL; ++t) {
    int p = t & 1;
#if __has_builtin(__builtin_amdgcn_s_wait_asynccnt)
    __builtin_amdgcn_s_wait_asynccnt(0);
#else
    asm volatile("s_wait_asynccnt 0" ::: "memory");
#endif
    __syncthreads();                         // bc[p] visible to all waves
    if (t + 1 < SEQL && tid < 8) {           // prefetch t+1 into bc[p^1]
      const float* g = dbc + ((size_t)b * SEQL + t + 1) * DBCW + DRANK + tid * 4;
      unsigned lo = ((unsigned)(uintptr_t)&bc[p ^ 1][0]) + tid * 16;
      asm volatile("global_load_async_to_lds_b128 %0, %1, off" :: "v"(lo), "v"(g) : "memory");
    }
    long row = (long)b * SEQL + t;
    float dt = dta[row * EDIM + e];
    float ut = u[row * EDIM + e];
    float du = dt * ut, yt = 0.f;
    const float* Bs = &bc[p][0];
    const float* Cs = &bc[p][NSTATE];
#pragma unroll
    for (int n = 0; n < NSTATE; ++n) {
      h[n] = __expf(dt * A[n]) * h[n] + du * Bs[n];
      yt  += h[n] * Cs[n];
    }
    y[row * EDIM + e] = yt + ut * dpar;
  }
}

__global__ void k_gate(float* __restrict__ y, __bf16* __restrict__ yb,
                       const float* __restrict__ xz, long n) {
  long i = (long)blockIdx.x * blockDim.x + threadIdx.x;
  if (i >= n) return;
  int  e   = (int)(i % EDIM);
  long row = i / EDIM;
  float z = xz[row * (2 * EDIM) + EDIM + e];
  float g = z / (1.f + __expf(-z));
  float v = y[i] * g;
  y[i] = v; yb[i] = (__bf16)v;
}

__global__ void k_add(float* __restrict__ x, const float* __restrict__ o, long n) {
  long i = (long)blockIdx.x * blockDim.x + threadIdx.x;
  if (i < n) x[i] += o[i];
}

__global__ __launch_bounds__(256)
void k_head(const float* __restrict__ xn, const float* __restrict__ g,
            const float* __restrict__ bb, const float* __restrict__ dw,
            const float* __restrict__ db, float* __restrict__ nli) {
  __shared__ float red[256];
  int b = blockIdx.x;
  const float* cls = xn + (size_t)b * SEQL * DHID;
  float s = 0.f;
  for (int i = threadIdx.x; i < DHID; i += 256) s += cls[i];
  red[threadIdx.x] = s; __syncthreads();
  for (int st = 128; st; st >>= 1) { if (threadIdx.x < st) red[threadIdx.x] += red[threadIdx.x + st]; __syncthreads(); }
  float mu = red[0] / (float)DHID; __syncthreads();
  s = 0.f;
  for (int i = threadIdx.x; i < DHID; i += 256) { float d = cls[i] - mu; s += d * d; }
  red[threadIdx.x] = s; __syncthreads();
  for (int st = 128; st; st >>= 1) { if (threadIdx.x < st) red[threadIdx.x] += red[threadIdx.x + st]; __syncthreads(); }
  float inv = rsqrtf(red[0] / (float)DHID + 1e-12f); __syncthreads();
  for (int c = 0; c < NLAB; ++c) {
    s = 0.f;
    for (int i = threadIdx.x; i < DHID; i += 256)
      s += ((cls[i] - mu) * inv * g[i] + bb[i]) * dw[c * DHID + i];
    red[threadIdx.x] = s; __syncthreads();
    for (int st = 128; st; st >>= 1) { if (threadIdx.x < st) red[threadIdx.x] += red[threadIdx.x + st]; __syncthreads(); }
    if (threadIdx.x == 0) nli[b * NLAB + c] = red[0] + db[c];
    __syncthreads();
  }
}

__global__ void k_loss(const float* __restrict__ nli, const int* __restrict__ lab,
                       float* __restrict__ loss) {
  if (threadIdx.x != 0 || blockIdx.x != 0) return;
  float acc = 0.f;
  for (int b = 0; b < BSZ; ++b) {
    float m = -1e30f;
    for (int c = 0; c < NLAB; ++c) m = fmaxf(m, nli[b * NLAB + c]);
    float se = 0.f;
    for (int c = 0; c < NLAB; ++c) se += __expf(nli[b * NLAB + c] - m);
    int l = lab[b];
    acc -= (nli[b * NLAB + l] - m - __logf(se));
  }
  loss[0] = acc / (float)BSZ;
}

// ---------------- driver ----------------
extern "C" void kernel_launch(void* const* d_in, const int* in_sizes, int n_in,
                              void* d_out, int out_size, void* d_ws, size_t ws_size,
                              hipStream_t stream) {
  (void)in_sizes; (void)n_in; (void)out_size; (void)ws_size;
  const int*   tokens     = (const int*)  d_in[0];
  const int*   labels     = (const int*)  d_in[1];
  const float* emb        = (const float*)d_in[2];
  const float* rms_w      = (const float*)d_in[3];
  const float* in_proj_w  = (const float*)d_in[4];
  const float* conv_w     = (const float*)d_in[5];
  const float* conv_b     = (const float*)d_in[6];
  const float* x_proj_w   = (const float*)d_in[7];
  const float* dt_proj_w  = (const float*)d_in[8];
  const float* dt_proj_b  = (const float*)d_in[9];
  const float* A_log      = (const float*)d_in[10];
  const float* D_param    = (const float*)d_in[11];
  const float* out_proj_w = (const float*)d_in[12];
  const float* norm_f_w   = (const float*)d_in[13];
  const float* ln_g       = (const float*)d_in[14];
  const float* ln_b       = (const float*)d_in[15];
  const float* dense_w    = (const float*)d_in[16];
  const float* dense_b    = (const float*)d_in[17];

  char* ws = (char*)d_ws;
  size_t off = 0;
  auto alloc = [&](size_t bytes) { size_t o = off; off += (bytes + 255) & ~(size_t)255; return o; };
  const size_t M = MROWS;
  float*  x    = (float*) (ws + alloc(M * DHID * 4));
  __bf16* hbf  = (__bf16*)(ws + alloc(M * DHID * 2));
  float*  xz   = (float*) (ws + alloc(M * 2 * EDIM * 4));
  float*  xxs  = (float*) (ws + alloc(M * EDIM * 4));
  __bf16* xxsb = (__bf16*)(ws + alloc(M * EDIM * 2));
  float*  dbc  = (float*) (ws + alloc(M * DBCW * 4));
  __bf16* dlb  = (__bf16*)(ws + alloc(M * DRPAD * 2));
  float*  dta  = (float*) (ws + alloc(M * EDIM * 4));
  float*  ybuf = (float*) (ws + alloc(M * EDIM * 4));
  __bf16* ybb  = (__bf16*)(ws + alloc(M * EDIM * 2));
  float*  otmp = (float*) (ws + alloc(M * DHID * 4));
  __bf16* wInB = (__bf16*)(ws + alloc((size_t)2 * EDIM * DHID * 2));
  __bf16* wXB  = (__bf16*)(ws + alloc((size_t)DBCW * EDIM * 2));
  __bf16* wDtB = (__bf16*)(ws + alloc((size_t)EDIM * DRPAD * 2));
  __bf16* wOutB= (__bf16*)(ws + alloc((size_t)DHID * EDIM * 2));

  float* nli  = (float*)d_out;
  float* loss = (float*)d_out + BSZ * NLAB;
  float* xout = (float*)d_out + BSZ * NLAB + 1;

  const int T = 256;
  auto nb = [](long n) { return (int)((n + 255) / 256); };

  k_embed<<<nb((long)M * DHID), T, 0, stream>>>(tokens, emb, x, (long)M * DHID);

  for (int l = 0; l < NLAYER; ++l) {
    const float* rw  = rms_w      + (size_t)l * DHID;
    const float* ipw = in_proj_w  + (size_t)l * 2 * EDIM * DHID;
    const float* cw  = conv_w     + (size_t)l * EDIM * DCONV;
    const float* cb  = conv_b     + (size_t)l * EDIM;
    const float* xpw = x_proj_w   + (size_t)l * DBCW * EDIM;
    const float* dtw = dt_proj_w  + (size_t)l * EDIM * DRANK;
    const float* dtb = dt_proj_b  + (size_t)l * EDIM;
    const float* al  = A_log      + (size_t)l * EDIM * NSTATE;
    const float* dp  = D_param    + (size_t)l * EDIM;
    const float* opw = out_proj_w + (size_t)l * DHID * EDIM;

    // stage weights -> bf16
    k_f32_to_bf16<<<nb((long)2 * EDIM * DHID), T, 0, stream>>>(ipw, wInB, (long)2 * EDIM * DHID);
    k_f32_to_bf16<<<nb((long)DBCW * EDIM), T, 0, stream>>>(xpw, wXB, (long)DBCW * EDIM);
    k_cvt_pad<<<nb((long)EDIM * DRPAD), T, 0, stream>>>(dtw, wDtB, EDIM, DRANK, DRANK, DRPAD);
    k_f32_to_bf16<<<nb((long)DHID * EDIM), T, 0, stream>>>(opw, wOutB, (long)DHID * EDIM);

    // rmsnorm -> bf16 activations
    k_rmsnorm<<<(int)M, T, 0, stream>>>(x, rw, hbf, (float*)nullptr);

    // xz = h @ in_proj_w^T : [M,3072]
    k_gemm<4><<<dim3(2 * EDIM / 64, (int)(M / 128)), T, 0, stream>>>(hbf, wInB, xz, (int)M, 2 * EDIM, DHID);

    // causal conv + SiLU
    k_conv_silu<<<nb((long)M * EDIM), T, 0, stream>>>(xz, cw, cb, xxs, xxsb, (long)M * EDIM);

    // dbc = xx @ x_proj_w^T : [M,80] (NT=5 covers all 80 cols)
    k_gemm<5><<<dim3(1, (int)(M / 128)), T, 0, stream>>>(xxsb, wXB, dbc, (int)M, DBCW, EDIM);

    // d_low (first 48 cols of dbc) -> bf16 padded to 64
    k_cvt_pad<<<nb((long)M * DRPAD), T, 0, stream>>>(dbc, dlb, (long)M, DBCW, DRANK, DRPAD);

    // delta_raw = d_low @ dt_proj_w^T : [M,1536], then softplus(+bias)
    k_gemm<4><<<dim3(EDIM / 64, (int)(M / 128)), T, 0, stream>>>(dlb, wDtB, dta, (int)M, EDIM, DRPAD);
    k_softplus_bias<<<nb((long)M * EDIM), T, 0, stream>>>(dta, dtb, (long)M * EDIM);

    // selective scan (async double-buffered B/C broadcast)
    k_scan<<<dim3(EDIM / 256, BSZ), T, 0, stream>>>(xxs, dta, dbc, al, dp, ybuf);

    // gate with SiLU(z)
    k_gate<<<nb((long)M * EDIM), T, 0, stream>>>(ybuf, ybb, xz, (long)M * EDIM);

    // out_proj + residual
    k_gemm<4><<<dim3(DHID / 64, (int)(M / 128)), T, 0, stream>>>(ybb, wOutB, otmp, (int)M, DHID, EDIM);
    k_add<<<nb((long)M * DHID), T, 0, stream>>>(x, otmp, (long)M * DHID);
  }

  // final rmsnorm -> d_out x-section (f32)
  k_rmsnorm<<<(int)M, T, 0, stream>>>(x, norm_f_w, (__bf16*)nullptr, xout);

  // classification head + loss
  k_head<<<BSZ, T, 0, stream>>>(xout, ln_g, ln_b, dense_w, dense_b, nli);
  k_loss<<<1, 32, 0, stream>>>(nli, labels, loss);
}
